// BiDAF_11776800326158
// MI455X (gfx1250) — matrix operand order repaired and checked by reading, not compiled, not run on hardware
//
#include <hip/hip_runtime.h>
#include <hip/hip_bf16.h>

// ---------------------------------------------------------------------------
// BiDAF forward for MI455X (gfx1250, wave32, WMMA).
// - All dense GEMMs use v_wmma_f32_16x16x32_f16, 2 M-tiles per wave (B reuse),
//   global_prefetch on the A stream.
// - LSTM recurrences: ONE launch per layer with grid=2 (fwd+bwd concurrently,
//   one WGP each); Whh + biases staged in LDS; h/c/gates in LDS (~248KB/WGP).
// ---------------------------------------------------------------------------

typedef __attribute__((ext_vector_type(16))) _Float16 v16h;
typedef __attribute__((ext_vector_type(8)))  _Float16 v8h;
typedef __attribute__((ext_vector_type(8)))  float    v8f;

#define Bsz 64
#define TC  512
#define TQ  64
#define HID 100
#define D2  200     // 2*HID
#define KP200 224   // 200 padded to mult of 32
#define KP800 832   // 800 padded
#define NPAD208 208 // 200 padded to mult of 16 (N dim)

static inline int cdiv_i(long a, long b) { return (int)((a + b - 1) / b); }

__device__ __forceinline__ v8f wmma16(v16h a, v16h b, v8f c) {
  return __builtin_amdgcn_wmma_f32_16x16x32_f16(false, a, false, b, (short)0, c,
                                                false, false);
}

// ---------------------------------------------------------------------------
// Elementwise / small kernels
// ---------------------------------------------------------------------------

__global__ void char_cnn_k(const int* __restrict__ tok, const float* __restrict__ cemb,
                           const float* __restrict__ cw, const float* __restrict__ cb,
                           float* __restrict__ out, long NS) {
  long idx = (long)blockIdx.x * blockDim.x + threadIdx.x;
  if (idx >= NS * 100) return;
  long w = idx / 100;
  int ch = (int)(idx % 100);
  const int* tw = tok + w * 16;
  float best = 0.f;  // relu then max over positions -> >= 0
  for (int p = 0; p < 12; ++p) {
    float acc = cb[ch];
    for (int k = 0; k < 5; ++k) {
      const float* e = cemb + (long)tw[p + k] * 8;
      for (int c = 0; c < 8; ++c) acc += e[c] * cw[ch * 40 + c * 5 + k];
    }
    best = fmaxf(best, acc);
  }
  out[idx] = best;
}

__global__ void build_x16_k(const int* __restrict__ tok, const float* __restrict__ wemb,
                            const float* __restrict__ chf, _Float16* __restrict__ dst, long NS) {
  long idx = (long)blockIdx.x * blockDim.x + threadIdx.x;
  if (idx >= NS * KP200) return;
  long r = idx / KP200;
  int c = (int)(idx % KP200);
  float v = 0.f;
  if (c < 100)      v = wemb[(long)tok[r] * 100 + c];
  else if (c < 200) v = chf[r * 100 + (c - 100)];
  dst[idx] = (_Float16)v;
}

__global__ void cvt_pad_k(const float* __restrict__ src, _Float16* __restrict__ dst,
                          long rows, int cols, int lds, int kpad) {
  long idx = (long)blockIdx.x * blockDim.x + threadIdx.x;
  if (idx >= rows * kpad) return;
  long r = idx / kpad;
  int c = (int)(idx % kpad);
  dst[idx] = (c < cols) ? (_Float16)src[r * lds + c] : (_Float16)0.f;
}

__global__ void cvt_scale_pad_k(const float* __restrict__ src, const float* __restrict__ vec,
                                _Float16* __restrict__ dst, long rows, int cols, int lds, int kpad) {
  long idx = (long)blockIdx.x * blockDim.x + threadIdx.x;
  if (idx >= rows * kpad) return;
  long r = idx / kpad;
  int c = (int)(idx % kpad);
  dst[idx] = (c < cols) ? (_Float16)(src[r * lds + c] * vec[c]) : (_Float16)0.f;
}

// Q (B,64,200) -> Qt16 (B,208,64)
__global__ void qt_cvt_k(const float* __restrict__ Q, _Float16* __restrict__ Qt) {
  long idx = (long)blockIdx.x * blockDim.x + threadIdx.x;
  long total = (long)Bsz * NPAD208 * 64;
  if (idx >= total) return;
  int b = (int)(idx / (NPAD208 * 64));
  int d = (int)((idx / 64) % NPAD208);
  int j = (int)(idx % 64);
  Qt[idx] = (d < 200) ? (_Float16)Q[((long)b * 64 + j) * 200 + d] : (_Float16)0.f;
}

__global__ void dot_vec_k(const float* __restrict__ X, const float* __restrict__ w,
                          float* __restrict__ out, long rows, int D) {
  long r = (long)blockIdx.x * blockDim.x + threadIdx.x;
  if (r >= rows) return;
  const float* x = X + r * D;
  float s = 0.f;
  for (int d = 0; d < D; ++d) s += x[d] * w[d];
  out[r] = s;
}

__global__ void s_softmax_rows_k(float* __restrict__ S, const float* __restrict__ Cwc,
                                 const float* __restrict__ Qwq, float* __restrict__ Smax) {
  long r = (long)blockIdx.x * blockDim.x + threadIdx.x;
  if (r >= (long)Bsz * TC) return;
  int b = (int)(r >> 9);
  float* row = S + r * 64;
  const float* qb = Qwq + b * 64;
  float cv = Cwc[r];
  float v[64];
  float mx = -1e30f;
  for (int j = 0; j < 64; ++j) { v[j] = row[j] + cv + qb[j]; mx = fmaxf(mx, v[j]); }
  Smax[r] = mx;
  float s = 0.f;
  for (int j = 0; j < 64; ++j) { float e = __expf(v[j] - mx); v[j] = e; s += e; }
  float inv = 1.f / s;
  for (int j = 0; j < 64; ++j) row[j] = v[j] * inv;
}

__global__ void b_softmax_k(const float* __restrict__ Smax, float* __restrict__ bprob) {
  int b = threadIdx.x;
  if (b >= Bsz) return;
  const float* s = Smax + (long)b * TC;
  float mx = -1e30f;
  for (int i = 0; i < TC; ++i) mx = fmaxf(mx, s[i]);
  float sum = 0.f;
  for (int i = 0; i < TC; ++i) { float e = __expf(s[i] - mx); bprob[(long)b * TC + i] = e; sum += e; }
  float inv = 1.f / sum;
  for (int i = 0; i < TC; ++i) bprob[(long)b * TC + i] *= inv;
}

__global__ void q2c_k(const float* __restrict__ bprob, const float* __restrict__ C,
                      float* __restrict__ q2c) {
  long idx = (long)blockIdx.x * blockDim.x + threadIdx.x;
  if (idx >= (long)Bsz * D2) return;
  int b = (int)(idx / D2);
  int d = (int)(idx % D2);
  float s = 0.f;
  for (int i = 0; i < TC; ++i)
    s += bprob[(long)b * TC + i] * C[((long)b * TC + i) * D2 + d];
  q2c[idx] = s;
}

__global__ void build_G_k(const float* __restrict__ C, const float* __restrict__ c2q_pad,
                          const float* __restrict__ q2c, float* __restrict__ G) {
  long idx = (long)blockIdx.x * blockDim.x + threadIdx.x;
  if (idx >= (long)Bsz * TC * D2) return;
  long r = idx / D2;
  int d = (int)(idx % D2);
  int b = (int)(r >> 9);
  float cv = C[r * D2 + d];
  float c2 = c2q_pad[r * NPAD208 + d];
  float* g = G + r * 800;
  g[d]       = cv;
  g[200 + d] = c2;
  g[400 + d] = cv * c2;
  g[600 + d] = cv * q2c[(long)b * D2 + d];
}

__global__ void proj_k(const float* __restrict__ G, const float* __restrict__ M,
                       const float* __restrict__ w, float* __restrict__ out, long rows) {
  long r = (long)blockIdx.x * blockDim.x + threadIdx.x;
  if (r >= rows) return;
  const float* g = G + r * 800;
  const float* m = M + r * 200;
  float s = 0.f;
  for (int d = 0; d < 800; ++d) s += g[d] * w[d];
  for (int d = 0; d < 200; ++d) s += m[d] * w[800 + d];
  out[r] = s;
}

// ---------------------------------------------------------------------------
// WMMA GEMM: C[M,N] f32 = A[M,K]f16 @ W[N,K]f16^T, batched via blockIdx.y.
// Each wave computes a 32x16 tile: two A fragments share one B fragment
// (M must be a multiple of 32, N of 16, K of 32).
// ---------------------------------------------------------------------------
__global__ void gemm_wmma_k(const _Float16* __restrict__ A, const _Float16* __restrict__ Wt,
                            float* __restrict__ Cc, int M, int N, int K,
                            int lda, int ldw, int ldc,
                            long sA, long sW, long sC) {
  int lane = threadIdx.x & 31;
  int wave = threadIdx.x >> 5;
  int bz = blockIdx.y;
  const _Float16* Ab = A + (long)bz * sA;
  const _Float16* Wb = Wt + (long)bz * sW;
  float* Cb = Cc + (long)bz * sC;
  int tilesN = N >> 4;
  long tile = (long)blockIdx.x * 8 + wave;
  long ntiles = (long)(M >> 5) * tilesN;
  if (tile >= ntiles) return;  // wave-uniform
  int m0 = (int)(tile / tilesN) << 5;
  int n0 = (int)(tile % tilesN) << 4;

  int aoff = (lane & 16) ? 16 : 0;  // A K sub-offset (contiguous k = 16h + s, same order as B)
  int boff = (lane & 16) ? 16 : 0;  // B K sub-offset
  const _Float16* ap0 = Ab + (long)(m0 + (lane & 15)) * lda + aoff;
  const _Float16* ap1 = ap0 + (long)16 * lda;
  const _Float16* bp  = Wb + (long)(n0 + (lane & 15)) * ldw + boff;

  v8f acc0 = {}, acc1 = {};
  for (int k0 = 0; k0 < K; k0 += 32) {
    __builtin_prefetch(ap0 + k0 + 32, 0, 0);  // global_prefetch_b8
    __builtin_prefetch(ap1 + k0 + 32, 0, 0);
    v8h lo0 = *(const v8h*)(ap0 + k0);
    v8h hi0 = *(const v8h*)(ap0 + k0 + 8);
    v8h lo1 = *(const v8h*)(ap1 + k0);
    v8h hi1 = *(const v8h*)(ap1 + k0 + 8);
    v16h b = *(const v16h*)(bp + k0);
    v16h a0, a1;
    for (int i = 0; i < 8; ++i) {
      a0[i] = lo0[i]; a0[i + 8] = hi0[i];
      a1[i] = lo1[i]; a1[i + 8] = hi1[i];
    }
    acc0 = wmma16(a0, b, acc0);
    acc1 = wmma16(a1, b, acc1);
  }
  int rowb = m0 + ((lane & 16) ? 8 : 0);
  int col = n0 + (lane & 15);
  for (int r = 0; r < 8; ++r) {
    Cb[(long)(rowb + r) * ldc + col]      = acc0[r];
    Cb[(long)(rowb + 16 + r) * ldc + col] = acc1[r];
  }
}

// ---------------------------------------------------------------------------
// Bidirectional persistent LSTM: grid=2 (blockIdx.x = direction), 1024 thr
// (32 waves) per workgroup. LDS: Whh (400x128 f16), h (64x128 f16),
// c (64x100 f32), gates (64x400 f32), bias sum (400 f32) = ~248 KB / WGP.
// Per step: gates = h @ Whh^T (100 WMMA tiles over 32 waves) + xs + bias,
// then the LSTM cell update.
// ---------------------------------------------------------------------------
#define LSTM_SMEM (400 * 128 * 2 + 64 * 128 * 2 + 64 * 100 * 4 + 64 * 400 * 4 + 400 * 4)

__global__ void lstm_bidir_k(const float* __restrict__ xs_f, const float* __restrict__ xs_b,
                             const float* __restrict__ bih_f, const float* __restrict__ bhh_f,
                             const float* __restrict__ bih_b, const float* __restrict__ bhh_b,
                             const _Float16* __restrict__ whh_f, const _Float16* __restrict__ whh_b,
                             float* __restrict__ out, int T) {
  extern __shared__ char smem[];
  _Float16* whh  = (_Float16*)smem;                                        // 400*128
  _Float16* h16  = (_Float16*)(smem + 400 * 128 * 2);                      // 64*128
  float* c       = (float*)(smem + 400 * 128 * 2 + 64 * 128 * 2);          // 64*100
  float* gates   = (float*)(smem + 400 * 128 * 2 + 64 * 128 * 2 + 64 * 100 * 4); // 64*400
  float* bsum    = (float*)(smem + 400 * 128 * 2 + 64 * 128 * 2 + 64 * 100 * 4 + 64 * 400 * 4);

  int dir = blockIdx.x;  // 0 = forward, 1 = backward
  const float* xs  = dir ? xs_b : xs_f;
  const float* bih = dir ? bih_b : bih_f;
  const float* bhh = dir ? bhh_b : bhh_f;
  const _Float16* whhG = dir ? whh_b : whh_f;
  float* o = out + dir * HID;
  int reverse = dir;

  int tid = threadIdx.x;
  int lane = tid & 31, wave = tid >> 5;

  // stage Whh into LDS (v8h chunks), zero h/c, build bias sum
  for (int i = tid; i < 400 * 128 / 8; i += 1024)
    ((v8h*)whh)[i] = ((const v8h*)whhG)[i];
  for (int i = tid; i < 64 * 128; i += 1024) h16[i] = (_Float16)0.f;
  for (int i = tid; i < 64 * 100; i += 1024) c[i] = 0.f;
  for (int i = tid; i < 400; i += 1024) bsum[i] = bih[i] + bhh[i];
  __syncthreads();

  for (int t = 0; t < T; ++t) {
    int tt = reverse ? (T - 1 - t) : t;
    // phase 1: gates = h @ Whh^T + xs[:,tt,:] + bias
    for (int tile = wave; tile < 100; tile += 32) {  // wave-uniform
      int m0 = (tile / 25) << 4;
      int n0 = (tile % 25) << 4;
      int row = m0 + (lane & 15);
      int nb = n0 + (lane & 15);
      int aoff = (lane & 16) ? 16 : 0;
      int boff = (lane & 16) ? 16 : 0;
      v8f acc = {};
      for (int k0 = 0; k0 < 128; k0 += 32) {
        v8h lo = *(const v8h*)&h16[row * 128 + k0 + aoff];   // ds_load_b128
        v8h hi = *(const v8h*)&h16[row * 128 + k0 + aoff + 8];
        v16h a;
        for (int i = 0; i < 8; ++i) { a[i] = lo[i]; a[i + 8] = hi[i]; }
        v16h b = *(const v16h*)&whh[nb * 128 + k0 + boff];   // LDS B fragment
        acc = wmma16(a, b, acc);
      }
      int rowb = m0 + ((lane & 16) ? 8 : 0);
      int col = n0 + (lane & 15);
      for (int r = 0; r < 8; ++r) {
        int bi = rowb + r;
        gates[bi * 400 + col] = acc[r] + xs[((long)bi * T + tt) * 400 + col] + bsum[col];
      }
    }
    __syncthreads();
    // phase 2: cell update (i,f,g,o chunks of 100)
    for (int idx = tid; idx < 64 * 100; idx += 1024) {
      int bi = idx / 100, j = idx % 100;
      float gi = gates[bi * 400 + j];
      float gf = gates[bi * 400 + 100 + j];
      float gg = gates[bi * 400 + 200 + j];
      float go = gates[bi * 400 + 300 + j];
      float si = 1.f / (1.f + __expf(-gi));
      float sf = 1.f / (1.f + __expf(-gf));
      float so = 1.f / (1.f + __expf(-go));
      float cv = sf * c[idx] + si * tanhf(gg);
      c[idx] = cv;
      float hv = so * tanhf(cv);
      h16[bi * 128 + j] = (_Float16)hv;
      o[((long)bi * T + tt) * D2 + j] = hv;
    }
    __syncthreads();
  }
}

// ---------------------------------------------------------------------------
// Host-side helpers
// ---------------------------------------------------------------------------
static void run_bilstm(hipStream_t s, const float* P, int in_dim, int kpad,
                       const _Float16* X16, long rows, int T,
                       float* gf, float* gb, _Float16* wih16,
                       _Float16* whhf, _Float16* whhb, float* out) {
  // param layout per direction: Wih(400*in), Whh(400*100), bih(400), bhh(400)
  long blk = 400L * in_dim + 40000L + 800L;
  const float* Wih_f = P;
  const float* Whh_f = P + 400L * in_dim;
  const float* bih_f = Whh_f + 40000;
  const float* bhh_f = bih_f + 400;
  const float* Wih_b = P + blk;
  const float* Whh_b = Wih_b + 400L * in_dim;
  const float* bih_b = Whh_b + 40000;
  const float* bhh_b = bih_b + 400;

  long tiles = (rows / 32) * (400 / 16);
  int gx = cdiv_i(tiles, 8);

  cvt_pad_k<<<cdiv_i(400L * kpad, 256), 256, 0, s>>>(Wih_f, wih16, 400, in_dim, in_dim, kpad);
  gemm_wmma_k<<<dim3(gx, 1), 256, 0, s>>>(X16, wih16, gf, (int)rows, 400, kpad,
                                          kpad, kpad, 400, 0, 0, 0);
  cvt_pad_k<<<cdiv_i(400L * kpad, 256), 256, 0, s>>>(Wih_b, wih16, 400, in_dim, in_dim, kpad);
  gemm_wmma_k<<<dim3(gx, 1), 256, 0, s>>>(X16, wih16, gb, (int)rows, 400, kpad,
                                          kpad, kpad, 400, 0, 0, 0);
  cvt_pad_k<<<cdiv_i(400L * 128, 256), 256, 0, s>>>(Whh_f, whhf, 400, 100, 100, 128);
  cvt_pad_k<<<cdiv_i(400L * 128, 256), 256, 0, s>>>(Whh_b, whhb, 400, 100, 100, 128);
  // both directions in one launch -> two WGPs in parallel
  lstm_bidir_k<<<dim3(2), 1024, LSTM_SMEM, s>>>(gf, gb, bih_f, bhh_f, bih_b, bhh_b,
                                                whhf, whhb, out, T);
}

extern "C" void kernel_launch(void* const* d_in, const int* in_sizes, int n_in,
                              void* d_out, int out_size, void* d_ws, size_t ws_size,
                              hipStream_t stream) {
  const int* ctx    = (const int*)d_in[0];
  const int* ques   = (const int*)d_in[1];
  const int* cctx   = (const int*)d_in[2];
  const int* cques  = (const int*)d_in[3];
  const float* wemb = (const float*)d_in[4];
  const float* cemb = (const float*)d_in[5];
  const float* convw= (const float*)d_in[6];
  const float* convb= (const float*)d_in[7];
  const float* ctxl = (const float*)d_in[8];   // ctx_lstm params (f then b)
  const float* simw = (const float*)d_in[9];
  const float* ml1  = (const float*)d_in[10];
  const float* ml2  = (const float*)d_in[11];
  const float* endl = (const float*)d_in[12];
  const float* wst  = (const float*)d_in[13];
  const float* wen  = (const float*)d_in[14];
  float* out = (float*)d_out;

  // ---- workspace carve ----
  char* wp = (char*)d_ws;
  auto alloc = [&](size_t bytes) -> void* {
    void* p = wp;
    wp += (bytes + 255) & ~(size_t)255;
    return p;
  };
  const long RC = (long)Bsz * TC;   // 32768
  const long RQ = (long)Bsz * TQ;   // 4096
  float*     chf_c  = (float*)alloc(RC * 100 * 4);
  float*     chf_q  = (float*)alloc(RQ * 100 * 4);
  _Float16*  x16    = (_Float16*)alloc(RC * KP800 * 2);   // reused staging
  _Float16*  q16    = (_Float16*)alloc(RQ * KP200 * 2);
  _Float16*  qt16   = (_Float16*)alloc((long)Bsz * NPAD208 * 64 * 2);
  float*     gf     = (float*)alloc(RC * 400 * 4);
  float*     gb     = (float*)alloc(RC * 400 * 4);
  _Float16*  wih16  = (_Float16*)alloc(400L * KP800 * 2);
  _Float16*  whhf16 = (_Float16*)alloc(400L * 128 * 2);
  _Float16*  whhb16 = (_Float16*)alloc(400L * 128 * 2);
  float*     Cbuf   = (float*)alloc(RC * D2 * 4);
  float*     Qbuf   = (float*)alloc(RQ * D2 * 4);
  float*     L1out  = (float*)alloc(RC * D2 * 4);
  float*     Mbuf   = (float*)alloc(RC * D2 * 4);
  float*     M2buf  = (float*)alloc(RC * D2 * 4);
  float*     S      = (float*)alloc(RC * 64 * 4);
  float*     Smax   = (float*)alloc(RC * 4);
  float*     bprob  = (float*)alloc(RC * 4);
  float*     Cwc    = (float*)alloc(RC * 4);
  float*     Qwq    = (float*)alloc(RQ * 4);
  float*     c2qp   = (float*)alloc(RC * NPAD208 * 4);
  float*     q2c    = (float*)alloc((long)Bsz * D2 * 4);
  float*     G      = (float*)alloc(RC * 800 * 4);
  (void)in_sizes; (void)n_in; (void)out_size; (void)ws_size;

  // ---- 1. char CNN features ----
  char_cnn_k<<<cdiv_i(RC * 100, 256), 256, 0, stream>>>(cctx, cemb, convw, convb, chf_c, RC);
  char_cnn_k<<<cdiv_i(RQ * 100, 256), 256, 0, stream>>>(cques, cemb, convw, convb, chf_q, RQ);

  // ---- 2. context encoder BiLSTM -> C ----
  build_x16_k<<<cdiv_i(RC * KP200, 256), 256, 0, stream>>>(ctx, wemb, chf_c, x16, RC);
  run_bilstm(stream, ctxl, 200, KP200, x16, RC, TC, gf, gb, wih16, whhf16, whhb16, Cbuf);

  // ---- 3. question encoder BiLSTM -> Q ----
  build_x16_k<<<cdiv_i(RQ * KP200, 256), 256, 0, stream>>>(ques, wemb, chf_q, x16, RQ);
  run_bilstm(stream, ctxl, 200, KP200, x16, RQ, TQ, gf, gb, wih16, whhf16, whhb16, Qbuf);

  // ---- 4. attention: S = Cwc + Qwq + (C*w_cq) @ Q^T ----
  dot_vec_k<<<cdiv_i(RC, 256), 256, 0, stream>>>(Cbuf, simw, Cwc, RC, D2);
  dot_vec_k<<<cdiv_i(RQ, 256), 256, 0, stream>>>(Qbuf, simw + 200, Qwq, RQ, D2);
  cvt_scale_pad_k<<<cdiv_i(RC * KP200, 256), 256, 0, stream>>>(Cbuf, simw + 400, x16,
                                                               RC, D2, D2, KP200);
  cvt_pad_k<<<cdiv_i(RQ * KP200, 256), 256, 0, stream>>>(Qbuf, q16, RQ, D2, D2, KP200);
  {  // batched: per b, (512x224)@(224x64)
    long tiles = (TC / 32) * (64 / 16);
    gemm_wmma_k<<<dim3(cdiv_i(tiles, 8), Bsz), 256, 0, stream>>>(
        x16, q16, S, TC, 64, KP200, KP200, KP200, 64,
        (long)TC * KP200, (long)TQ * KP200, (long)TC * 64);
  }
  s_softmax_rows_k<<<cdiv_i(RC, 256), 256, 0, stream>>>(S, Cwc, Qwq, Smax);
  b_softmax_k<<<1, 64, 0, stream>>>(Smax, bprob);

  // ---- 5. c2q = a @ Q (batched WMMA, N padded 200->208) ----
  qt_cvt_k<<<cdiv_i((long)Bsz * NPAD208 * 64, 256), 256, 0, stream>>>(Qbuf, qt16);
  cvt_pad_k<<<cdiv_i(RC * 64, 256), 256, 0, stream>>>(S, x16, RC, 64, 64, 64);  // a -> f16
  {
    long tiles = (TC / 32) * (NPAD208 / 16);
    gemm_wmma_k<<<dim3(cdiv_i(tiles, 8), Bsz), 256, 0, stream>>>(
        x16, qt16, c2qp, TC, NPAD208, 64, 64, 64, NPAD208,
        (long)TC * 64, (long)NPAD208 * 64, (long)TC * NPAD208);
  }
  q2c_k<<<cdiv_i((long)Bsz * D2, 256), 256, 0, stream>>>(bprob, Cbuf, q2c);

  // ---- 6. G = [C | c2q | C*c2q | C*q2c] ----
  build_G_k<<<cdiv_i(RC * D2, 256), 256, 0, stream>>>(Cbuf, c2qp, q2c, G);

  // ---- 7. modeling stacks ----
  cvt_pad_k<<<cdiv_i(RC * KP800, 256), 256, 0, stream>>>(G, x16, RC, 800, 800, KP800);
  run_bilstm(stream, ml1, 800, KP800, x16, RC, TC, gf, gb, wih16, whhf16, whhb16, L1out);
  cvt_pad_k<<<cdiv_i(RC * KP200, 256), 256, 0, stream>>>(L1out, x16, RC, D2, D2, KP200);
  run_bilstm(stream, ml2, 200, KP200, x16, RC, TC, gf, gb, wih16, whhf16, whhb16, Mbuf);
  cvt_pad_k<<<cdiv_i(RC * KP200, 256), 256, 0, stream>>>(Mbuf, x16, RC, D2, D2, KP200);
  run_bilstm(stream, endl, 200, KP200, x16, RC, TC, gf, gb, wih16, whhf16, whhb16, M2buf);

  // ---- 8. span projections ----
  proj_k<<<cdiv_i(RC, 256), 256, 0, stream>>>(G, Mbuf, wst, out, RC);
  proj_k<<<cdiv_i(RC, 256), 256, 0, stream>>>(G, M2buf, wen, out + RC, RC);
}
